// LowerPrecisionBigNet_28063316312490
// MI455X (gfx1250) — matrix operand order repaired and checked by reading, not compiled, not run on hardware
//
#include <hip/hip_runtime.h>
#include <hip/hip_bf16.h>
#include <stdint.h>

// ---------------------------------------------------------------------------
// Fused 5-block quantized-MLP for MI455X (gfx1250, wave32, WMMA).
// One workgroup (256 threads = 8 waves) owns a 32-row batch tile and runs all
// 15 linears + 4 layernorms with activations resident in LDS.
//   - GEMM: v_wmma_f32_16x16x32_f16, per wave: 2 M-tiles x 8 N-tiles; every
//     weight is dequantized exactly once per workgroup.
//   - Dequant = 1 VALU op/weight: fp16 magic 0x6400|q == 1024+q via
//     v_perm_b32, then v_pk_fma_f16 with per-group (s/15, -1024*s/15) LUT.
//   - Weight/LUT loads: loop-carried base pointers + constant instruction
//     offsets, clause-batched in groups of 4 N-tiles.
//   - amdgpu_num_vgpr(256): keep allocation inside the direct-address VGPR
//     window so the scheduler throttles pipelining instead of emitting
//     s_set_vgpr_msb mode switches through the hot loop.
//   - LDS row stride padded to 1032 halves: A-fragment lanes hit disjoint
//     bank groups; ds_load_b128 feeds WMMA operands directly.
// ---------------------------------------------------------------------------

#define C_DIM    1024
#define MT       32             // batch rows per workgroup
#define STR      1032           // padded LDS row stride (halves)
#define N_LAYERS 15
#define LN_EPS   1e-5f

typedef _Float16 v16h __attribute__((ext_vector_type(16)));
typedef _Float16 v8h  __attribute__((ext_vector_type(8)));
typedef _Float16 v4h  __attribute__((ext_vector_type(4)));
typedef _Float16 h2   __attribute__((ext_vector_type(2)));
typedef float    v8f  __attribute__((ext_vector_type(8)));

// ---------------------------------------------------------------------------
// GEMM over one 32x1024 tile: aOut = act(aIn @ W^T + bias [+ res]).
// PACKED=true : weights are 1 byte/elem in wq8, scales pre-baked in lut.
// PACKED=false: fallback, weights are int32 boxed, scales fp16 in wn.
// ---------------------------------------------------------------------------
template<bool PACKED>
__device__ __forceinline__ void gemm_tile(
    const _Float16* __restrict__ aIn, _Float16* __restrict__ aOut,
    const _Float16* __restrict__ res,              // nullptr if no residual
    const uint8_t*  __restrict__ wq8,              // [1024][1024] bytes
    const uint2*    __restrict__ lutL,             // [65536] {(s,s),(c,c)}
    const int*      __restrict__ wq32,             // fallback weights
    const _Float16* __restrict__ wnL,              // fallback scales
    const float*    __restrict__ biasL,
    bool doRelu)
{
  const int lane    = threadIdx.x & 31;
  const int wave    = threadIdx.x >> 5;
  const int nlo     = lane & 15;
  const int hi      = lane >> 4;
  const int colBase = wave << 7;                   // 128 output cols per wave
  const int nbase   = colBase + nlo;               // weight row for nt==0

  // Loop-invariant per-lane base pointers; nt indexing is a constant offset.
  const uint8_t* qrow = wq8  + (size_t)nbase * C_DIM + hi * 16;
  const uint2*   lrow = lutL + nbase * 64 + hi;

  v8f acc[2][8];
#pragma unroll
  for (int mt = 0; mt < 2; ++mt)
#pragma unroll
    for (int nt = 0; nt < 8; ++nt)
#pragma unroll
      for (int e = 0; e < 8; ++e)
        acc[mt][nt][e] = 0.0f;

  for (int kt = 0; kt < C_DIM; kt += 32) {
    // ---- A fragments (2 M-tiles) from LDS: two ds_load_b128 each ----------
    const _Float16* pa = aIn + nlo * STR + kt + hi * 16;
    v8h a0lo = *(const v8h*)(pa);
    v8h a0hi = *(const v8h*)(pa + 8);
    v8h a1lo = *(const v8h*)(pa + 16 * STR);
    v8h a1hi = *(const v8h*)(pa + 16 * STR + 8);
    v16h a0, a1;
#pragma unroll
    for (int i = 0; i < 8; ++i) {
      a0[i] = a0lo[i]; a0[8 + i] = a0hi[i];
      a1[i] = a1lo[i]; a1[8 + i] = a1hi[i];
    }

    const uint8_t* qk   = qrow + kt;               // bumps by 32 per k-step
    const uint2*   lk   = lrow + (kt >> 4);        // bumps by 2 per k-step
    const int      krow = kt + hi * 16;            // this lane's K start

    // Two groups of 4 N-tiles: 4x b128 + 4x b64 loads in flight per group.
#pragma unroll
    for (int gsel = 0; gsel < 2; ++gsel) {
      uint4 wv[4];
      uint2 scv[4];
      if (PACKED) {
#pragma unroll
        for (int j = 0; j < 4; ++j) {
          const int nt = gsel * 4 + j;
          wv[j]  = *(const uint4*)(qk + nt * 16384); // offset:nt*16384
          scv[j] = lk[nt * 1024];                    // offset:nt*8192
        }
      }
#pragma unroll
      for (int j = 0; j < 4; ++j) {
        const int nt = gsel * 4 + j;
        v16h b;
        if (PACKED) {
          const h2 s2 = __builtin_bit_cast(h2, scv[j].x);
          const h2 c2 = __builtin_bit_cast(h2, scv[j].y);
          const uint32_t dw[4] = {wv[j].x, wv[j].y, wv[j].z, wv[j].w};
#pragma unroll
          for (int p = 0; p < 4; ++p) {
            // pair = (1024+q0, 1024+q1) as fp16 via byte permute with 0x64.
            uint32_t p0 = __builtin_amdgcn_perm(0x64646464u, dw[p], 0x04010400u);
            uint32_t p1 = __builtin_amdgcn_perm(0x64646464u, dw[p], 0x04030402u);
            h2 w0 = __builtin_elementwise_fma(__builtin_bit_cast(h2, p0), s2, c2);
            h2 w1 = __builtin_elementwise_fma(__builtin_bit_cast(h2, p1), s2, c2);
            b[4 * p + 0] = w0[0]; b[4 * p + 1] = w0[1];
            b[4 * p + 2] = w1[0]; b[4 * p + 3] = w1[1];
          }
        } else {
          const int n = colBase + (nt << 4) + nlo;
          const int* q = wq32 + (size_t)n * C_DIM + krow;
          const float s = (float)wnL[(n << 6) + (krow >> 4)] * (1.0f / 15.0f);
#pragma unroll
          for (int i = 0; i < 16; ++i)
            b[i] = (_Float16)((float)q[i] * s);
        }
        acc[0][nt] = __builtin_amdgcn_wmma_f32_16x16x32_f16(
            false, a0, false, b, (short)0, acc[0][nt], false, false);
        acc[1][nt] = __builtin_amdgcn_wmma_f32_16x16x32_f16(
            false, a1, false, b, (short)0, acc[1][nt], false, false);
      }
    }
  }

  // ---- fused epilogue: bias (+residual) (+relu), fp32 -> fp16 LDS ---------
#pragma unroll
  for (int nt = 0; nt < 8; ++nt) {
    const int col = colBase + (nt << 4) + nlo;
    const float bv = biasL[col];
#pragma unroll
    for (int mt = 0; mt < 2; ++mt) {
#pragma unroll
      for (int r = 0; r < 8; ++r) {
        const int m = (mt << 4) + (hi << 3) + r;   // C/D layout: M = r + 8*hi
        float v = acc[mt][nt][r] + bv;
        if (res)    v += (float)res[m * STR + col];
        if (doRelu) v = fmaxf(v, 0.0f);
        aOut[m * STR + col] = (_Float16)v;
      }
    }
  }
}

// Row-wise layernorm; one wave handles 4 rows, single fused sum/sumsq pass
// with vectorized b64 LDS accesses, then shfl_xor wave reductions.
__device__ __forceinline__ void layernorm_tile(
    _Float16* __restrict__ buf,
    const float* __restrict__ g, const float* __restrict__ beta)
{
  const int lane = threadIdx.x & 31;
  const int wave = threadIdx.x >> 5;
#pragma unroll
  for (int i = 0; i < 4; ++i) {
    _Float16* row = buf + (wave * 4 + i) * STR;
    float sum = 0.0f, sq = 0.0f;
    for (int c = lane * 4; c < C_DIM; c += 128) {
      v4h h = *(const v4h*)(row + c);
#pragma unroll
      for (int e = 0; e < 4; ++e) {
        float v = (float)h[e];
        sum += v; sq += v * v;
      }
    }
#pragma unroll
    for (int off = 16; off; off >>= 1) {
      sum += __shfl_xor(sum, off, 32);
      sq  += __shfl_xor(sq,  off, 32);
    }
    const float mu  = sum * (1.0f / C_DIM);
    const float var = sq * (1.0f / C_DIM) - mu * mu;
    const float inv = rsqrtf(var + LN_EPS);

    for (int c = lane * 4; c < C_DIM; c += 128) {
      v4h h = *(const v4h*)(row + c);
      v4h o;
#pragma unroll
      for (int e = 0; e < 4; ++e)
        o[e] = (_Float16)(((float)h[e] - mu) * inv * g[c + e] + beta[c + e]);
      *(v4h*)(row + c) = o;
    }
  }
}

template<bool PACKED>
__global__ void __launch_bounds__(256, 4)
__attribute__((amdgpu_num_vgpr(256)))
bignet_kernel(const float* __restrict__ x,
              const uint8_t* __restrict__ wq8,
              const uint2*   __restrict__ lut,
              const int*     __restrict__ wq32,
              const _Float16* __restrict__ wn,
              const float* __restrict__ bias,
              const float* __restrict__ ln_g,
              const float* __restrict__ ln_b,
              float* __restrict__ out)
{
  extern __shared__ __align__(16) _Float16 smem[];
  _Float16* bufA = smem;                       // 32 x 1032 halves
  _Float16* bufB = smem + MT * STR;
  _Float16* bres = smem + 2 * MT * STR;

  const int tid  = threadIdx.x;
  const int row0 = blockIdx.x * MT;

  // load + fp32->fp16 the input tile (b128 global reads, b64 LDS writes)
  for (int idx = tid; idx < MT * C_DIM / 4; idx += 256) {
    const int r = idx >> 8, c4 = (idx & 255) * 4;
    const float4 v = *(const float4*)(x + (size_t)(row0 + r) * C_DIM + c4);
    v4h h = {(_Float16)v.x, (_Float16)v.y, (_Float16)v.z, (_Float16)v.w};
    *(v4h*)(bufA + r * STR + c4) = h;
  }
  __syncthreads();

  _Float16* cur = bufA;
  _Float16* alt = bufB;

  for (int blk = 0; blk < 5; ++blk) {
    // residual snapshot: raw b128 copy of the whole padded buffer
    {
      const uint4* s = (const uint4*)cur;
      uint4*       d = (uint4*)bres;
      for (int idx = tid; idx < MT * STR / 8; idx += 256) d[idx] = s[idx];
    }
    __syncthreads();

    const int i = blk * 3;
    gemm_tile<PACKED>(cur, alt, (const _Float16*)nullptr,
                      wq8  + (size_t)i * C_DIM * C_DIM,
                      lut  + (size_t)i * 65536,
                      wq32 + (size_t)i * C_DIM * C_DIM,
                      wn   + (size_t)i * 65536,
                      bias + i * C_DIM, true);
    __syncthreads();
    gemm_tile<PACKED>(alt, cur, (const _Float16*)nullptr,
                      wq8  + (size_t)(i + 1) * C_DIM * C_DIM,
                      lut  + (size_t)(i + 1) * 65536,
                      wq32 + (size_t)(i + 1) * C_DIM * C_DIM,
                      wn   + (size_t)(i + 1) * 65536,
                      bias + (i + 1) * C_DIM, true);
    __syncthreads();
    gemm_tile<PACKED>(cur, alt, bres,
                      wq8  + (size_t)(i + 2) * C_DIM * C_DIM,
                      lut  + (size_t)(i + 2) * 65536,
                      wq32 + (size_t)(i + 2) * C_DIM * C_DIM,
                      wn   + (size_t)(i + 2) * 65536,
                      bias + (i + 2) * C_DIM, false);
    __syncthreads();

    _Float16* t = cur; cur = alt; alt = t;
    if (blk < 4) {
      layernorm_tile(cur, ln_g + blk * C_DIM, ln_b + blk * C_DIM);
      __syncthreads();
    }
  }

  // fp16 -> fp32 output (b64 LDS reads, b128 global writes)
  for (int idx = tid; idx < MT * C_DIM / 4; idx += 256) {
    const int r = idx >> 8, c4 = (idx & 255) * 4;
    v4h h = *(const v4h*)(cur + r * STR + c4);
    float4 v = make_float4((float)h[0], (float)h[1], (float)h[2], (float)h[3]);
    *(float4*)(out + (size_t)(row0 + r) * C_DIM + c4) = v;
  }
}

// ---------------------------------------------------------------------------
// Pre-pass 1: repack int32-boxed weights to 1 byte/elem (1/4 the L2 traffic).
__global__ void __launch_bounds__(256)
pack_wq_kernel(const int4* __restrict__ src, uchar4* __restrict__ dst, int n4)
{
  const int i = blockIdx.x * blockDim.x + threadIdx.x;
  if (i < n4) {
    int4 v = src[i];
    dst[i] = make_uchar4((unsigned char)v.x, (unsigned char)v.y,
                         (unsigned char)v.z, (unsigned char)v.w);
  }
}

// Pre-pass 2: per-group fp16 constants for the magic dequant:
//   spair = (s/15, s/15), cpair = (-1024*s/15, -1024*s/15)
__global__ void __launch_bounds__(256)
build_lut_kernel(const _Float16* __restrict__ wn, uint2* __restrict__ lut, int n)
{
  const int i = blockIdx.x * blockDim.x + threadIdx.x;
  if (i < n) {
    const float s = (float)wn[i] * (1.0f / 15.0f);
    const _Float16 hs = (_Float16)s;
    const _Float16 hc = (_Float16)(-1024.0f * (float)hs);
    const uint32_t us = (uint32_t)__builtin_bit_cast(unsigned short, hs);
    const uint32_t uc = (uint32_t)__builtin_bit_cast(unsigned short, hc);
    lut[i] = make_uint2(us | (us << 16), uc | (uc << 16));
  }
}

extern "C" void kernel_launch(void* const* d_in, const int* in_sizes, int n_in,
                              void* d_out, int out_size, void* d_ws, size_t ws_size,
                              hipStream_t stream) {
  const float*    x    = (const float*)d_in[0];
  const int*      wq32 = (const int*)d_in[1];     // int weights, 0..15
  const _Float16* wn   = (const _Float16*)d_in[2];
  const float*    bias = (const float*)d_in[3];
  const float*    ln_g = (const float*)d_in[4];
  const float*    ln_b = (const float*)d_in[5];
  float*          out  = (float*)d_out;

  const int rows = in_sizes[0] / C_DIM;           // 8192
  const int grid = rows / MT;                     // 256 workgroups
  const size_t lds_bytes = (size_t)3 * MT * STR * sizeof(_Float16); // ~194 KB

  const size_t wq_elems  = (size_t)N_LAYERS * C_DIM * C_DIM;        // 15.7 M
  const size_t n_groups  = (size_t)N_LAYERS * 65536;                // 983 K
  const size_t lut_bytes = n_groups * sizeof(uint2);                // 7.9 MB
  const size_t ws_need   = wq_elems + lut_bytes;                    // 23.6 MB

  if (ws_size >= ws_need) {
    uint8_t* wq8 = (uint8_t*)d_ws;
    uint2*   lut = (uint2*)((uint8_t*)d_ws + wq_elems);

    const int n4 = (int)(wq_elems / 4);
    pack_wq_kernel<<<(n4 + 255) / 256, 256, 0, stream>>>(
        (const int4*)wq32, (uchar4*)wq8, n4);
    build_lut_kernel<<<((int)n_groups + 255) / 256, 256, 0, stream>>>(
        wn, lut, (int)n_groups);

    (void)hipFuncSetAttribute(
        reinterpret_cast<const void*>(&bignet_kernel<true>),
        hipFuncAttributeMaxDynamicSharedMemorySize, (int)lds_bytes);
    bignet_kernel<true><<<grid, 256, lds_bytes, stream>>>(
        x, wq8, lut, wq32, wn, bias, ln_g, ln_b, out);
  } else {
    (void)hipFuncSetAttribute(
        reinterpret_cast<const void*>(&bignet_kernel<false>),
        hipFuncAttributeMaxDynamicSharedMemorySize, (int)lds_bytes);
    bignet_kernel<false><<<grid, 256, lds_bytes, stream>>>(
        x, (const uint8_t*)nullptr, (const uint2*)nullptr,
        wq32, wn, bias, ln_g, ln_b, out);
  }
}
